// WindowAttention_33251636805781
// MI455X (gfx1250) — matrix operand order repaired and checked by reading, not compiled, not run on hardware
//
#include <hip/hip_runtime.h>
#include <hip/hip_bf16.h>

// ---- problem constants (from reference) -----------------------------------
#define NWIN   49              // window tokens (7x7)
#define HEADS  8
#define HD     32              // head dim
#define CDIM   256
#define BWIN   4096            // number of windows
#define MROWS  (BWIN * NWIN)   // 200704 flattened rows
#define NMASK  64

typedef _Float16 h16 __attribute__((ext_vector_type(16)));
typedef _Float16 h8  __attribute__((ext_vector_type(8)));
typedef float    f8  __attribute__((ext_vector_type(8)));
typedef float    f4  __attribute__((ext_vector_type(4)));

static __device__ __forceinline__ f8 wmma_f16(h16 a, h16 b, f8 c) {
  // v_wmma_f32_16x16x32_f16  (neg_a, A, neg_b, B, c_mod, C, reuse_a, reuse_b)
  return __builtin_amdgcn_wmma_f32_16x16x32_f16(false, a, false, b, (short)0, c,
                                                false, false);
}

static __device__ __forceinline__ h16 cat8(h8 lo, h8 hi) {
  return __builtin_shufflevector(lo, hi, 0, 1, 2, 3, 4, 5, 6, 7,
                                 8, 9, 10, 11, 12, 13, 14, 15);
}

// ---------------------------------------------------------------------------
// Kernel: f32 -> f16 bulk convert (8 elements / thread, b128 in, b128 out)
// ---------------------------------------------------------------------------
__global__ __launch_bounds__(256) void cvt_f16_kernel(
    const float* __restrict__ in, _Float16* __restrict__ out, long n8) {
  const long i = ((long)blockIdx.x * blockDim.x + threadIdx.x);
  if (i >= n8) return;
  const f4 a = *(const f4*)(in + i * 8);
  const f4 b = *(const f4*)(in + i * 8 + 4);
  h8 o;
#pragma unroll
  for (int e = 0; e < 4; ++e) { o[e] = (_Float16)a[e]; o[e + 4] = (_Float16)b[e]; }
  *(h8*)(out + i * 8) = o;
}

// ---------------------------------------------------------------------------
// Kernel 0: relative-position bias gather  bias[h][i][j] = table[idx[i][j]][h]
// ---------------------------------------------------------------------------
__global__ __launch_bounds__(256) void bias_kernel(
    const float* __restrict__ rel_table, const int* __restrict__ rel_index,
    float* __restrict__ bias_ws) {
  const int h = blockIdx.x;
  for (int idx = threadIdx.x; idx < NWIN * NWIN; idx += blockDim.x)
    bias_ws[h * NWIN * NWIN + idx] = rel_table[rel_index[idx] * HEADS + h];
}

// ---------------------------------------------------------------------------
// Kernel 1: QKV projection GEMM (200704x256) x (256x768), f16 WMMA.
// Inputs pre-converted to f16 (x_h L2-resident across the 12 N-tile passes).
// Epilogue scatters into q (scaled), k (row-major N x hd) and vT (hd x 64).
// ---------------------------------------------------------------------------
__global__ __launch_bounds__(128) void qkv_kernel(
    const _Float16* __restrict__ x_h, const _Float16* __restrict__ w_h,
    const float* __restrict__ qkv_b,
    _Float16* __restrict__ q_ws, _Float16* __restrict__ k_ws,
    _Float16* __restrict__ vT_ws) {
  const int lane = threadIdx.x & 31;
  const int wv   = threadIdx.x >> 5;
  const int l16  = lane & 15;
  const int half = lane >> 4;
  const long m0  = (long)blockIdx.x * 64 + wv * 16;
  const int  n0  = blockIdx.y * 64;

  f8 acc[4] = {};
  const _Float16* ar = x_h + (m0 + l16) * (long)CDIM;
  const _Float16* br0 = w_h + (long)(n0 + l16) * CDIM + half * 16;

#pragma unroll
  for (int k0 = 0; k0 < CDIM; k0 += 32) {
    const h16 a = *(const h16*)(ar + k0 + half * 16);
#pragma unroll
    for (int ct = 0; ct < 4; ++ct) {
      const h16 b = *(const h16*)(br0 + (long)ct * 16 * CDIM + k0);
      acc[ct] = wmma_f16(a, b, acc[ct]);
    }
  }

  const float qscale = 0.17677669529663688f;  // hd^-0.5
#pragma unroll
  for (int ct = 0; ct < 4; ++ct) {
    const int ncol = n0 + ct * 16 + l16;       // column in [0,768)
    const float bv = qkv_b[ncol];
    const int which = ncol >> 8;               // 0=q 1=k 2=v
    const int cc = ncol & 255;
    const int h = cc >> 5;
    const int d = cc & 31;
#pragma unroll
    for (int r = 0; r < 8; ++r) {
      const long m = m0 + r + half * 8;        // flat row
      const int b_ = (int)(m / NWIN);
      const int n_ = (int)(m % NWIN);
      const float v = acc[ct][r] + bv;
      if (which == 0)
        q_ws[((long)(b_ * HEADS + h) * NWIN + n_) * HD + d] = (_Float16)(v * qscale);
      else if (which == 1)
        k_ws[((long)(b_ * HEADS + h) * NWIN + n_) * HD + d] = (_Float16)v;
      else
        vT_ws[((long)(b_ * HEADS + h) * HD + d) * 64 + n_] = (_Float16)v;
    }
  }
}

// ---------------------------------------------------------------------------
// Kernel 2: per-(window, head) attention.  One block = 128 threads = 4 waves;
// wave wv owns attention rows [wv*16, wv*16+16).  N=49 padded to 64.
//   S = q k^T  (4 WMMA)  -> +bias +mask -> LDS -> softmax -> f16 probs
//   O = P v    (2 Kchunks x 2 hd tiles = 4 WMMA)
// ---------------------------------------------------------------------------
__global__ __launch_bounds__(128) void attn_kernel(
    const _Float16* __restrict__ q_ws, const _Float16* __restrict__ k_ws,
    const _Float16* __restrict__ vT_ws, const float* __restrict__ bias_ws,
    const float* __restrict__ mask, _Float16* __restrict__ attn_out) {
  __shared__ float    sAttn[4][16][64];
  __shared__ _Float16 sProb[4][16][64];

  const int lane = threadIdx.x & 31;
  const int wv   = threadIdx.x >> 5;
  const int l16  = lane & 15;
  const int half = lane >> 4;

  const int bh = blockIdx.x;
  const int b_ = bh >> 3;
  const int h  = bh & 7;
  const int w  = b_ & (NMASK - 1);

  const _Float16* qb = q_ws + (long)bh * NWIN * HD;
  const _Float16* kb = k_ws + (long)bh * NWIN * HD;
  const _Float16* vb = vT_ws + (long)bh * HD * 64;
  const float* biash = bias_ws + (long)h * NWIN * NWIN;
  const float* maskw = mask + (long)w * NWIN * NWIN;

  // ---- A fragment: this wave's 16 q rows (zero pad beyond 49) -------------
  h16 a = {};
  {
    const int row = wv * 16 + l16;
    if (row < NWIN) {
      const _Float16* qr = qb + row * HD;
      a = *(const h16*)(qr + half * 16);
    }
  }

  // ---- S = q k^T, one 16x16 tile per j ------------------------------------
#pragma unroll
  for (int j = 0; j < 4; ++j) {
    h16 bf = {};
    const int krow = j * 16 + l16;
    if (krow < NWIN) bf = *(const h16*)(kb + krow * HD + half * 16);
    f8 c = {};
    c = wmma_f16(a, bf, c);
    const int ncol = j * 16 + l16;
#pragma unroll
    for (int r = 0; r < 8; ++r) {
      const int mloc = r + half * 8;
      const int roww = wv * 16 + mloc;
      float v = -1e30f;
      if (roww < NWIN && ncol < NWIN)
        v = c[r] + biash[roww * NWIN + ncol] + maskw[roww * NWIN + ncol];
      sAttn[wv][mloc][ncol] = v;
    }
  }

  __syncthreads();

  // ---- row softmax: one lane per row (lanes 0..15) ------------------------
  if (lane < 16) {
    float* rowp  = sAttn[wv][lane];
    _Float16* pp = sProb[wv][lane];
    float mx = -1e30f;
    for (int j = 0; j < NWIN; ++j) mx = fmaxf(mx, rowp[j]);
    float s = 0.0f;
    for (int j = 0; j < NWIN; ++j) {
      const float e = __expf(rowp[j] - mx);
      s += e;
      rowp[j] = e;
    }
    const float inv = 1.0f / s;
    for (int j = 0; j < NWIN; ++j) pp[j] = (_Float16)(rowp[j] * inv);
    for (int j = NWIN; j < 64; ++j) pp[j] = (_Float16)0.0f;  // exact zero pad
  }

  __syncthreads();

  // ---- O = P v : (16x64) x (64x32), vT gives contiguous B fragments -------
  f8 o[2] = {};
#pragma unroll
  for (int kc = 0; kc < 2; ++kc) {
    const _Float16* pr = sProb[wv][l16] + kc * 32;
    const h16 ap = *(const h16*)(pr + half * 16);
#pragma unroll
    for (int ct = 0; ct < 2; ++ct) {
      const h16 bv = *(const h16*)(vb + (ct * 16 + l16) * 64 + kc * 32 + half * 16);
      o[ct] = wmma_f16(ap, bv, o[ct]);
    }
  }

#pragma unroll
  for (int ct = 0; ct < 2; ++ct) {
#pragma unroll
    for (int r = 0; r < 8; ++r) {
      const int mloc = r + half * 8;
      const int nwin = wv * 16 + mloc;
      if (nwin < NWIN)
        attn_out[((long)b_ * NWIN + nwin) * CDIM + h * HD + ct * 16 + l16] =
            (_Float16)o[ct][r];
    }
  }
}

// ---------------------------------------------------------------------------
// Kernel 3: output projection GEMM (200704x256) x (256x256) + bias -> f32 out
// ---------------------------------------------------------------------------
__global__ __launch_bounds__(128) void proj_kernel(
    const _Float16* __restrict__ ain, const _Float16* __restrict__ w_h,
    const float* __restrict__ proj_b, float* __restrict__ out) {
  const int lane = threadIdx.x & 31;
  const int wv   = threadIdx.x >> 5;
  const int l16  = lane & 15;
  const int half = lane >> 4;
  const long m0  = (long)blockIdx.x * 64 + wv * 16;
  const int  n0  = blockIdx.y * 64;

  f8 acc[4] = {};
  const _Float16* ar  = ain + (m0 + l16) * (long)CDIM;
  const _Float16* br0 = w_h + (long)(n0 + l16) * CDIM + half * 16;

#pragma unroll
  for (int k0 = 0; k0 < CDIM; k0 += 32) {
    const h16 a = *(const h16*)(ar + k0 + half * 16);
#pragma unroll
    for (int ct = 0; ct < 4; ++ct) {
      const h16 b = *(const h16*)(br0 + (long)ct * 16 * CDIM + k0);
      acc[ct] = wmma_f16(a, b, acc[ct]);
    }
  }

#pragma unroll
  for (int ct = 0; ct < 4; ++ct) {
    const int ncol = n0 + ct * 16 + l16;
    const float pb = proj_b[ncol];
#pragma unroll
    for (int r = 0; r < 8; ++r) {
      const long m = m0 + r + half * 8;
      out[m * CDIM + ncol] = acc[ct][r] + pb;
    }
  }
}

// ---------------------------------------------------------------------------
extern "C" void kernel_launch(void* const* d_in, const int* in_sizes, int n_in,
                              void* d_out, int out_size, void* d_ws, size_t ws_size,
                              hipStream_t stream) {
  const float* x        = (const float*)d_in[0];
  const float* mask     = (const float*)d_in[1];
  const float* qkv_w    = (const float*)d_in[2];
  const float* qkv_b    = (const float*)d_in[3];
  const float* rel_tab  = (const float*)d_in[4];
  const float* proj_w   = (const float*)d_in[5];
  const float* proj_b   = (const float*)d_in[6];
  const int*   rel_idx  = (const int*)d_in[7];
  float* out = (float*)d_out;

  char* ws = (char*)d_ws;
  size_t off = 0;
  auto carve = [&](size_t bytes) -> char* {
    char* p = ws + off;
    off = (off + bytes + 255) & ~(size_t)255;
    return p;
  };

  const size_t xh_elems  = (size_t)MROWS * CDIM;           // 51,380,224
  const size_t wq_elems  = (size_t)3 * CDIM * CDIM;        // 196,608
  const size_t wp_elems  = (size_t)CDIM * CDIM;            // 65,536
  const size_t qk_bytes  = (size_t)BWIN * HEADS * NWIN * HD * sizeof(_Float16);
  const size_t vt_bytes  = (size_t)BWIN * HEADS * HD * 64 * sizeof(_Float16);
  const size_t ao_bytes  = (size_t)MROWS * CDIM * sizeof(_Float16);
  const size_t bws_bytes = (size_t)HEADS * NWIN * NWIN * sizeof(float);

  _Float16* x_h     = (_Float16*)carve(xh_elems * sizeof(_Float16));
  _Float16* wqkv_h  = (_Float16*)carve(wq_elems * sizeof(_Float16));
  _Float16* wproj_h = (_Float16*)carve(wp_elems * sizeof(_Float16));
  _Float16* q_ws    = (_Float16*)carve(qk_bytes);
  _Float16* k_ws    = (_Float16*)carve(qk_bytes);
  _Float16* vT_ws   = (_Float16*)carve(vt_bytes);
  _Float16* aout    = (_Float16*)carve(ao_bytes);
  float*    bias_ws = (float*)carve(bws_bytes);

  // zero vT so padded columns (only multiplied by exact-zero probs) are clean
  hipMemsetAsync(vT_ws, 0, vt_bytes, stream);

  // one-time f32 -> f16 conversions (x_h stays L2-resident: 103 MB < 192 MB)
  cvt_f16_kernel<<<dim3((unsigned)((xh_elems / 8 + 255) / 256)), dim3(256), 0,
                   stream>>>(x, x_h, (long)(xh_elems / 8));
  cvt_f16_kernel<<<dim3((unsigned)((wq_elems / 8 + 255) / 256)), dim3(256), 0,
                   stream>>>(qkv_w, wqkv_h, (long)(wq_elems / 8));
  cvt_f16_kernel<<<dim3((unsigned)((wp_elems / 8 + 255) / 256)), dim3(256), 0,
                   stream>>>(proj_w, wproj_h, (long)(wp_elems / 8));

  bias_kernel<<<dim3(HEADS), dim3(256), 0, stream>>>(rel_tab, rel_idx, bias_ws);
  qkv_kernel<<<dim3(MROWS / 64, 12), dim3(128), 0, stream>>>(
      x_h, wqkv_h, qkv_b, q_ws, k_ws, vT_ws);
  attn_kernel<<<dim3(BWIN * HEADS), dim3(128), 0, stream>>>(
      q_ws, k_ws, vT_ws, bias_ws, mask, aout);
  proj_kernel<<<dim3(MROWS / 64, 4), dim3(128), 0, stream>>>(
      aout, wproj_h, proj_b, out);
}
